// Transformer_85942295592974
// MI455X (gfx1250) — matrix operand order repaired and checked by reading, not compiled, not run on hardware
//
#include <hip/hip_runtime.h>
#include <hip/hip_bf16.h>

// ---------------------------------------------------------------------------
// Transformer encoder-decoder forward for MI455X (gfx1250, wave32, WMMA).
// All heavy matmuls use v_wmma_f32_16x16x32_bf16 with fp32 accumulation.
// All WMMA operands are staged contiguous-bf16 so fragments load as b128s.
// ---------------------------------------------------------------------------

#define NLAYERS 6
#define DMODEL  512
#define NHEAD   8
#define DHEAD   64
#define DFF     2048
#define BSZ     4
#define SLEN    1024
#define NTOK    (BSZ * SLEN)          // 4096
#define SC_STRIDE 1032                // f32 scores row stride (8-bank skew)
#define PB_STRIDE 1040                // bf16 P row stride (16B-aligned, skewed)

typedef __attribute__((ext_vector_type(16))) __bf16 v16bf;
typedef __attribute__((ext_vector_type(8)))  __bf16 v8bf;
typedef __attribute__((ext_vector_type(8)))  float  v8f;

__device__ __forceinline__ v8f wmma_bf16(v16bf a, v16bf b, v8f c) {
    return __builtin_amdgcn_wmma_f32_16x16x32_bf16(false, a, false, b,
                                                   (short)0, c, false, false);
}

// A-fragment (16x32, ISA 7.12.2): lane hi*16+r16 holds row r16; elements 0..7
// are K-locals hi*8+0..7, elements 8..15 are 16+hi*8+0..7  -> two 16B loads.
__device__ __forceinline__ v16bf cat8(v8bf lo, v8bf hi8) {
    v16bf r;
#pragma unroll
    for (int j = 0; j < 8; ++j) { r[j] = lo[j]; r[8 + j] = hi8[j]; }
    return r;
}

// ---------------------------------------------------------------------------
// GEMM: C[M,N] = A[M,K](bf16) x WT[N,K](bf16, pre-transposed) + bias (+ReLU).
// Output fp32 (Cf) or bf16 (Cb).  256 thr = 8 waves; 64x64 block tile.
// ---------------------------------------------------------------------------
__global__ __launch_bounds__(256) void gemm_wmma_kernel(
    const __bf16* __restrict__ A, const __bf16* __restrict__ WT,
    float* __restrict__ Cf, __bf16* __restrict__ Cb,
    const float* __restrict__ bias, int M, int K, int N, int relu)
{
    const int tid  = threadIdx.x;
    const int wid  = tid >> 5;
    const int lane = tid & 31;
    const int r16  = lane & 15;
    const int hi   = lane >> 4;

    const int mtile  = blockIdx.y * 64 + (wid & 3) * 16;
    const int ntile0 = blockIdx.x * 64 + (wid >> 2) * 32;

    const long abase = (long)(mtile + r16) * K;
    const int  n0 = ntile0 + r16;
    const int  n1 = n0 + 16;
    const long wb0 = (long)n0 * K;
    const long wb1 = (long)n1 * K;

    v8f acc0 = {};
    v8f acc1 = {};

    const int nsteps = K >> 5;
    for (int s = 0; s < nsteps; ++s) {
        const int kbase = s << 5;
        __builtin_prefetch(WT + wb0 + kbase + 128, 0, 1);   // global_prefetch_b8

        const v16bf af = *(const v16bf*)(A + abase + kbase + (hi << 4));
        const v16bf b0 = *(const v16bf*)(WT + wb0 + kbase + (hi << 4));
        const v16bf b1 = *(const v16bf*)(WT + wb1 + kbase + (hi << 4));
        acc0 = wmma_bf16(af, b0, acc0);
        acc1 = wmma_bf16(af, b1, acc1);
    }

#pragma unroll
    for (int r = 0; r < 8; ++r) {
        const int  row = mtile + r + (hi << 3);
        const long rb  = (long)row * N;
        float v0 = acc0[r], v1 = acc1[r];
        if (bias) { v0 += bias[n0]; v1 += bias[n1]; }
        if (relu) { v0 = fmaxf(v0, 0.f); v1 = fmaxf(v1, 0.f); }
        if (Cf) { Cf[rb + n0] = v0;          Cf[rb + n1] = v1; }
        else    { Cb[rb + n0] = (__bf16)v0;  Cb[rb + n1] = (__bf16)v1; }
    }
}

// ---------------------------------------------------------------------------
// Weight transpose+convert: W fp32 (plain [K,N] or headed (H,K,64)) ->
// WT bf16 [N,K].  LDS-tiled 32x32 so both sides stay coalesced.
// ---------------------------------------------------------------------------
__global__ __launch_bounds__(256) void wt_kernel(
    const float* __restrict__ W, __bf16* __restrict__ WT,
    int K, int N, int headed)
{
    __shared__ float t[32][33];
    const int k0 = blockIdx.y * 32, n0 = blockIdx.x * 32;
    const int tx = threadIdx.x & 31, ty = threadIdx.x >> 5;
#pragma unroll
    for (int j = 0; j < 32; j += 8) {
        const int k = k0 + ty + j, n = n0 + tx;
        long src;
        if (headed) { const int h = n >> 6, kk = n & 63; src = ((long)h * K + k) * 64 + kk; }
        else        { src = (long)k * N + n; }
        t[ty + j][tx] = W[src];
    }
    __syncthreads();
#pragma unroll
    for (int j = 0; j < 32; j += 8)
        WT[(long)(n0 + ty + j) * K + k0 + tx] = (__bf16)t[tx][ty + j];
}

// V fp32 [B*S, 512] -> VT bf16 [B, 512, S]   (row = b*512 + h*64 + dh)
__global__ __launch_bounds__(256) void vt_kernel(
    const float* __restrict__ v, __bf16* __restrict__ VT)
{
    __shared__ float t[32][33];
    const int b  = blockIdx.z;
    const int c0 = blockIdx.x * 32;   // channel
    const int t0 = blockIdx.y * 32;   // token
    const int tx = threadIdx.x & 31, ty = threadIdx.x >> 5;
#pragma unroll
    for (int j = 0; j < 32; j += 8)
        t[ty + j][tx] = v[((long)b * SLEN + t0 + ty + j) * DMODEL + c0 + tx];
    __syncthreads();
#pragma unroll
    for (int j = 0; j < 32; j += 8)
        VT[((long)b * DMODEL + c0 + ty + j) * SLEN + t0 + tx] = (__bf16)t[tx][ty + j];
}

// fp32 -> bf16 elementwise
__global__ __launch_bounds__(256) void conv_kernel(
    const float* __restrict__ x, __bf16* __restrict__ y)
{
    const long i = (long)blockIdx.x * 256 + threadIdx.x;
    y[i] = (__bf16)x[i];
}

// ---------------------------------------------------------------------------
// Fused attention for one (b, h, 16-query tile).
// q,k bf16 [B,S,512]; vt bf16 [B,512,S]; o bf16 [B,S,512].
// ---------------------------------------------------------------------------
__global__ __launch_bounds__(128) void attn_wmma_kernel(
    const __bf16* __restrict__ q, const __bf16* __restrict__ k,
    const __bf16* __restrict__ vt, __bf16* __restrict__ o, int causal)
{
    __shared__ float  sc[16 * SC_STRIDE];   // raw scores (fp32)
    __shared__ __bf16 pb[16 * PB_STRIDE];   // normalized P (bf16)

    const int qt = blockIdx.x;
    const int h  = blockIdx.y;
    const int b  = blockIdx.z;
    const int qbase = qt * 16;

    const int tid  = threadIdx.x;
    const int wid  = tid >> 5;
    const int lane = tid & 31;
    const int r16  = lane & 15;
    const int hi   = lane >> 4;

    const long seq = (long)b * SLEN;
    const __bf16* qp = q + seq * DMODEL + h * DHEAD;
    const __bf16* kp = k + seq * DMODEL + h * DHEAD;

    // ---- Phase 1: scores S = Q K^T / 8 ------------------------------------
    v16bf qa0, qa1;
    {
        const long rb = (long)(qbase + r16) * DMODEL;
        qa0 = *(const v16bf*)(qp + rb + (hi << 4));
        qa1 = *(const v16bf*)(qp + rb + 32 + (hi << 4));
    }
    for (int kt = wid; kt < SLEN / 16; kt += 4) {
        const int ktb = kt * 16;
        if (causal && ktb > qbase + 15) continue;
        const long tb = (long)(ktb + r16) * DMODEL;
        const v16bf kb0 = *(const v16bf*)(kp + tb + (hi << 4));
        const v16bf kb1 = *(const v16bf*)(kp + tb + 32 + (hi << 4));
        v8f acc = {};
        acc = wmma_bf16(qa0, kb0, acc);
        acc = wmma_bf16(qa1, kb1, acc);
#pragma unroll
        for (int r = 0; r < 8; ++r)
            sc[(r + (hi << 3)) * SC_STRIDE + ktb + r16] = acc[r] * 0.125f;
    }
    __syncthreads();

    // ---- Phase 2: softmax -> bf16 P (8 lanes per row, intra-wave) ----------
    {
        const int row = tid >> 3;
        const int sub = tid & 7;
        const int lim = causal ? (qbase + row) : (SLEN - 1);
        float* srow = sc + row * SC_STRIDE;
        __bf16* prow = pb + row * PB_STRIDE;

        float mx = -1e30f;
        for (int c = sub; c <= lim; c += 8) mx = fmaxf(mx, srow[c]);
#pragma unroll
        for (int m = 1; m < 8; m <<= 1) mx = fmaxf(mx, __shfl_xor(mx, m, 32));

        float sum = 0.f;
        for (int c = sub; c < SLEN; c += 8) {
            float e = (c <= lim) ? __expf(srow[c] - mx) : 0.f;
            srow[c] = e;
            sum += e;
        }
#pragma unroll
        for (int m = 1; m < 8; m <<= 1) sum += __shfl_xor(sum, m, 32);

        const float inv = 1.f / sum;
        for (int c = sub; c < SLEN; c += 8) prow[c] = (__bf16)(srow[c] * inv);
    }
    __syncthreads();

    // ---- Phase 3: O = P V (wave owns a 16-wide dh slice) -------------------
    const int dhc = wid * 16 + r16;
    const __bf16* vrow = vt + ((long)b * DMODEL + h * DHEAD + dhc) * SLEN;
    v8f acc = {};
    const int kmax = causal ? (((qbase + 16) + 31) & ~31) : SLEN;
    for (int kb = 0; kb < kmax; kb += 32) {
        const v16bf pa = *(const v16bf*)(pb + r16 * PB_STRIDE + kb + (hi << 4));
        const v16bf vb = *(const v16bf*)(vrow + kb + (hi << 4));
        acc = wmma_bf16(pa, vb, acc);
    }
#pragma unroll
    for (int r = 0; r < 8; ++r) {
        const int row = qbase + r + (hi << 3);
        o[(seq + row) * DMODEL + h * DHEAD + dhc] = (__bf16)acc[r];
    }
}

// ---------------------------------------------------------------------------
// Fused residual + LayerNorm; emits fp32 master and (optional) bf16 copy.
// ---------------------------------------------------------------------------
__global__ __launch_bounds__(256) void ln_res_kernel(
    const float* __restrict__ x, const float* __restrict__ res,
    const float* __restrict__ g, const float* __restrict__ bb,
    float* __restrict__ out, __bf16* __restrict__ outb)
{
    const int  row  = blockIdx.x;
    const int  tid  = threadIdx.x;
    const long base = (long)row * DMODEL;

    const float t0 = x[base + tid]       + res[base + tid];
    const float t1 = x[base + tid + 256] + res[base + tid + 256];
    float s  = t0 + t1;
    float ss = t0 * t0 + t1 * t1;
#pragma unroll
    for (int m = 1; m < 32; m <<= 1) {
        s  += __shfl_xor(s,  m, 32);
        ss += __shfl_xor(ss, m, 32);
    }
    __shared__ float s1[8], s2[8];
    const int wid = tid >> 5, lane = tid & 31;
    if (lane == 0) { s1[wid] = s; s2[wid] = ss; }
    __syncthreads();
    float S = 0.f, SS = 0.f;
#pragma unroll
    for (int j = 0; j < 8; ++j) { S += s1[j]; SS += s2[j]; }
    const float mean = S * (1.f / DMODEL);
    const float var  = SS * (1.f / DMODEL) - mean * mean;
    const float inv  = rsqrtf(var + 1e-5f);
    const float y0 = g[tid]       * (t0 - mean) * inv + bb[tid];
    const float y1 = g[tid + 256] * (t1 - mean) * inv + bb[tid + 256];
    out[base + tid]       = y0;
    out[base + tid + 256] = y1;
    if (outb) { outb[base + tid] = (__bf16)y0; outb[base + tid + 256] = (__bf16)y1; }
}

// ---------------------------------------------------------------------------
// Host orchestration
// ---------------------------------------------------------------------------
extern "C" void kernel_launch(void* const* d_in, const int* in_sizes, int n_in,
                              void* d_out, int out_size, void* d_ws, size_t ws_size,
                              hipStream_t stream)
{
    (void)in_sizes; (void)n_in; (void)out_size; (void)ws_size;
#define IN(i) ((const float*)d_in[(i)])
    const float* enc_x = IN(0);
    const float* dec_x = IN(1);
    // d_in[182] = dec_mask (causal tril) - implemented analytically.

    char* ws = (char*)d_ws;
    const size_t MB = 1ull << 20;
    // fp32 masters
    float* X0 = (float*)(ws + 0 * MB);
    float* X1 = (float*)(ws + 8 * MB);
    float* X2 = (float*)(ws + 16 * MB);
    float* XT = (float*)(ws + 24 * MB);
    float* D0 = (float*)(ws + 32 * MB);
    float* D1 = (float*)(ws + 40 * MB);
    float* Vf = (float*)(ws + 48 * MB);
    float* P1 = (float*)(ws + 56 * MB);
    // bf16 staging
    __bf16* X0b  = (__bf16*)(ws + 64 * MB);
    __bf16* X1b  = (__bf16*)(ws + 68 * MB);
    __bf16* X2b  = (__bf16*)(ws + 72 * MB);
    __bf16* XTb  = (__bf16*)(ws + 76 * MB);
    __bf16* D0b  = (__bf16*)(ws + 80 * MB);
    __bf16* D1b  = (__bf16*)(ws + 84 * MB);
    __bf16* EXb  = (__bf16*)(ws + 88 * MB);
    __bf16* DXb  = (__bf16*)(ws + 92 * MB);
    __bf16* Qb   = (__bf16*)(ws + 96 * MB);
    __bf16* Kb   = (__bf16*)(ws + 100 * MB);
    __bf16* VT   = (__bf16*)(ws + 104 * MB);
    __bf16* AOb  = (__bf16*)(ws + 108 * MB);
    __bf16* MIDb = (__bf16*)(ws + 112 * MB);   // 16 MB
    __bf16* WT   = (__bf16*)(ws + 128 * MB);   //  2 MB  -> 130 MB total

    auto wt = [&](const float* W, int K, int N, int headed) {
        wt_kernel<<<dim3(N / 32, K / 32), 256, 0, stream>>>(W, WT, K, N, headed);
    };
    auto gemm = [&](const __bf16* A, float* Cf, __bf16* Cb, const float* bias,
                    int M, int K, int N, int relu) {
        gemm_wmma_kernel<<<dim3(N / 64, M / 64), 256, 0, stream>>>(
            A, WT, Cf, Cb, bias, M, K, N, relu);
    };
    auto attn = [&](const __bf16* q, const __bf16* k, const __bf16* v,
                    __bf16* o, int causal) {
        attn_wmma_kernel<<<dim3(SLEN / 16, NHEAD, BSZ), 128, 0, stream>>>(
            q, k, v, o, causal);
    };
    auto ln = [&](const float* x, const float* r, const float* g, const float* b,
                  float* o, __bf16* ob) {
        ln_res_kernel<<<NTOK, 256, 0, stream>>>(x, r, g, b, o, ob);
    };

    conv_kernel<<<(NTOK * DMODEL) / 256, 256, 0, stream>>>(enc_x, EXb);
    conv_kernel<<<(NTOK * DMODEL) / 256, 256, 0, stream>>>(dec_x, DXb);

    // ---------------- Encoder ----------------
    const float*  xin  = enc_x;
    const __bf16* xinb = EXb;
    for (int l = 0; l < NLAYERS; ++l) {
        const int p = 2 + l * 12;
        const float *Wq = IN(p+0), *Wk = IN(p+1), *Wv = IN(p+2), *Wo = IN(p+3);
        const float *W1 = IN(p+4), *b1 = IN(p+5), *W2 = IN(p+6), *b2 = IN(p+7);
        const float *g1 = IN(p+8), *be1 = IN(p+9), *g2 = IN(p+10), *be2 = IN(p+11);

        wt(Wq, DMODEL, DMODEL, 1); gemm(xinb, nullptr, Qb, nullptr, NTOK, DMODEL, DMODEL, 0);
        wt(Wk, DMODEL, DMODEL, 1); gemm(xinb, nullptr, Kb, nullptr, NTOK, DMODEL, DMODEL, 0);
        wt(Wv, DMODEL, DMODEL, 1); gemm(xinb, Vf, nullptr, nullptr, NTOK, DMODEL, DMODEL, 0);
        vt_kernel<<<dim3(16, 32, BSZ), 256, 0, stream>>>(Vf, VT);
        attn(Qb, Kb, VT, AOb, 0);
        wt(Wo, DMODEL, DMODEL, 0); gemm(AOb, P1, nullptr, nullptr, NTOK, DMODEL, DMODEL, 0);
        ln(P1, xin, g1, be1, X2, X2b);
        wt(W1, DMODEL, DFF, 0);    gemm(X2b, nullptr, MIDb, b1, NTOK, DMODEL, DFF, 1);
        wt(W2, DFF, DMODEL, 0);    gemm(MIDb, P1, nullptr, b2, NTOK, DFF, DMODEL, 0);
        float*  xo  = (l & 1) ? X0 : X1;
        __bf16* xob = (l & 1) ? X0b : X1b;
        ln(P1, X2, g2, be2, xo, xob);
        xin = xo; xinb = xob;
    }
    const __bf16* encb = xinb;   // == X0b after 6 layers

    // ---------------- Decoder ----------------
    const float*  din  = dec_x;
    const __bf16* dinb = DXb;
    for (int l = 0; l < NLAYERS; ++l) {
        const int p = 2 + NLAYERS * 12 + l * 18;
        const float *sWq = IN(p+0),  *sWk = IN(p+1),  *sWv = IN(p+2),  *sWo = IN(p+3);
        const float *cWq = IN(p+4),  *cWk = IN(p+5),  *cWv = IN(p+6),  *cWo = IN(p+7);
        const float *W1  = IN(p+8),  *b1  = IN(p+9),  *W2  = IN(p+10), *b2  = IN(p+11);
        const float *g1  = IN(p+12), *be1 = IN(p+13), *g2  = IN(p+14), *be2 = IN(p+15);
        const float *g3  = IN(p+16), *be3 = IN(p+17);

        // masked self-attention
        wt(sWq, DMODEL, DMODEL, 1); gemm(dinb, nullptr, Qb, nullptr, NTOK, DMODEL, DMODEL, 0);
        wt(sWk, DMODEL, DMODEL, 1); gemm(dinb, nullptr, Kb, nullptr, NTOK, DMODEL, DMODEL, 0);
        wt(sWv, DMODEL, DMODEL, 1); gemm(dinb, Vf, nullptr, nullptr, NTOK, DMODEL, DMODEL, 0);
        vt_kernel<<<dim3(16, 32, BSZ), 256, 0, stream>>>(Vf, VT);
        attn(Qb, Kb, VT, AOb, 1);
        wt(sWo, DMODEL, DMODEL, 0); gemm(AOb, P1, nullptr, nullptr, NTOK, DMODEL, DMODEL, 0);
        ln(P1, din, g1, be1, X2, X2b);

        // cross-attention (K/V from encoder output)
        wt(cWq, DMODEL, DMODEL, 1); gemm(X2b, nullptr, Qb, nullptr, NTOK, DMODEL, DMODEL, 0);
        wt(cWk, DMODEL, DMODEL, 1); gemm(encb, nullptr, Kb, nullptr, NTOK, DMODEL, DMODEL, 0);
        wt(cWv, DMODEL, DMODEL, 1); gemm(encb, Vf, nullptr, nullptr, NTOK, DMODEL, DMODEL, 0);
        vt_kernel<<<dim3(16, 32, BSZ), 256, 0, stream>>>(Vf, VT);
        attn(Qb, Kb, VT, AOb, 0);
        wt(cWo, DMODEL, DMODEL, 0); gemm(AOb, P1, nullptr, nullptr, NTOK, DMODEL, DMODEL, 0);
        ln(P1, X2, g2, be2, XT, XTb);

        // FFN
        wt(W1, DMODEL, DFF, 0); gemm(XTb, nullptr, MIDb, b1, NTOK, DMODEL, DFF, 1);
        wt(W2, DFF, DMODEL, 0); gemm(MIDb, P1, nullptr, b2, NTOK, DFF, DMODEL, 0);
        const bool last = (l == NLAYERS - 1);
        float*  dof = last ? (float*)d_out : ((l & 1) ? D0 : D1);
        __bf16* dob = last ? nullptr       : ((l & 1) ? D0b : D1b);
        ln(P1, XT, g3, be3, dof, dob);
        din = dof; dinb = dob;
    }
#undef IN
}
